// QuantumQLSTM_65481071395978
// MI455X (gfx1250) — matrix operand order repaired and checked by reading, not compiled, not run on hardware
//
#include <hip/hip_runtime.h>

typedef __attribute__((ext_vector_type(16))) _Float16 v16h;
typedef __attribute__((ext_vector_type(8)))  _Float16 v8h;
typedef __attribute__((ext_vector_type(8)))  float    v8f;
typedef __attribute__((ext_vector_type(4)))  float    v4f;

#define T_   2048
#define B_   512
#define D_   128
#define H_   128
#define DH_  256
#define HSTR 136   // LDS h-buffer row stride in halves (272B rows, bank-conflict free)
#define ZSTR 36    // LDS z-scratch row stride in floats (padded)

__device__ __forceinline__ float sigm(float x)      { return 1.0f / (1.0f + __expf(-x)); }
__device__ __forceinline__ float tanh_fast(float x) { return 1.0f - 2.0f / (__expf(2.0f * x) + 1.0f); }

// Build a v16h fragment from 16 f32 values (elements 0..15 <-> k order of the frag)
__device__ __forceinline__ v16h cvt_frag(v4f a, v4f b, v4f c, v4f d) {
  v16h r;
#pragma unroll
  for (int i = 0; i < 4; i++) {
    r[i]      = (_Float16)a[i];
    r[4 + i]  = (_Float16)b[i];
    r[8 + i]  = (_Float16)c[i];
    r[12 + i] = (_Float16)d[i];
  }
  return r;
}

__global__ __launch_bounds__(128, 1)
void qlstm_scan_kernel(const float* __restrict__ inp,
                       const float* __restrict__ Wf, const float* __restrict__ bf,
                       const float* __restrict__ Wi, const float* __restrict__ bi,
                       const float* __restrict__ Wu, const float* __restrict__ bu,
                       const float* __restrict__ Wo, const float* __restrict__ bo,
                       float* __restrict__ out)
{
  __shared__ _Float16 lds_h[16 * HSTR];  // h_{t-1} tile, f16, row-major [16][128] padded
  __shared__ float    zbuf[16 * ZSTR];   // sin^2 scratch [16 samples][32 gate-qubit]
  __shared__ float    gbuf[64];          // gate scalars: f[16], i[16], u[16], o[16]

  const int tid  = threadIdx.x;
  const int lane = tid & 31;
  const int m0   = blockIdx.x * 16;      // 32 blocks x 16 samples = B

  float* outStack = out;
  float* outHx    = out + (size_t)T_ * B_ * H_;
  float* outCx    = outHx + (size_t)B_ * H_;

  // zero the h buffer (h0 = 0)
  for (int i = tid; i < 16 * HSTR; i += 128) lds_h[i] = (_Float16)0.0f;

  // element-wise ownership: sample em, columns ej..ej+15 ; c state in registers (c0 = 0)
  const int em = tid >> 3;
  const int ej = (tid & 7) * 16;
  float creg[16];
#pragma unroll
  for (int i = 0; i < 16; i++) creg[i] = 0.0f;

  // ---- wave-0 resident state --------------------------------------------------
  v16h wbx[2][4], wbh[2][4];  // B-operand weight frags: [ntile][kchunk]
  v8f  xacc0, xacc1;          // pipelined x-projection accumulators (bias + x@Wx)
  float bias0 = 0.0f, bias1 = 0.0f;
  const int an  = lane & 15;             // A: sample row / B: output column / C: n
  const int klo = (lane < 16) ? 0 : 8;   // B-frag k base within 32-chunk
  const int koA = (lane < 16) ? 0 : 8;   // A-frag k offset within 32-chunk

  if (tid < 32) {
    const float* Ws[4] = {Wf, Wi, Wu, Wo};
    const float* bs[4] = {bf, bi, bu, bo};
#pragma unroll
    for (int nt = 0; nt < 2; nt++) {
      int g = nt * 16 + an;                         // gate-qubit output index 0..31
      const float* wrow = Ws[g >> 3] + (size_t)(g & 7) * DH_;
#pragma unroll
      for (int c = 0; c < 4; c++) {
        const float* px = wrow + c * 32 + klo;        // x part (cols 0..127)
        const float* ph = wrow + 128 + c * 32 + klo;  // h part (cols 128..255)
        wbx[nt][c] = cvt_frag(*(const v4f*)(px),      *(const v4f*)(px + 4),
                              *(const v4f*)(px + 16), *(const v4f*)(px + 20));
        wbh[nt][c] = cvt_frag(*(const v4f*)(ph),      *(const v4f*)(ph + 4),
                              *(const v4f*)(ph + 16), *(const v4f*)(ph + 20));
      }
      float bv = bs[g >> 3][g & 7];
      if (nt == 0) bias0 = bv; else bias1 = bv;
    }
    // x-projection for t = 0 computed ahead of the loop
    {
      const float* xr = inp + (size_t)(m0 + an) * D_;
      v16h xf[4];
#pragma unroll
      for (int c = 0; c < 4; c++) {
        const float* p0 = xr + c * 32 + koA;
        const float* p1 = xr + c * 32 + 16 + koA;
        xf[c] = cvt_frag(*(const v4f*)p0, *(const v4f*)(p0 + 4),
                         *(const v4f*)p1, *(const v4f*)(p1 + 4));
      }
#pragma unroll
      for (int v = 0; v < 8; v++) { xacc0[v] = bias0; xacc1[v] = bias1; }
#pragma unroll
      for (int c = 0; c < 4; c++) {
        xacc0 = __builtin_amdgcn_wmma_f32_16x16x32_f16(false, xf[c], false, wbx[0][c],
                                                       (short)0, xacc0, false, false);
        xacc1 = __builtin_amdgcn_wmma_f32_16x16x32_f16(false, xf[c], false, wbx[1][c],
                                                       (short)0, xacc1, false, false);
      }
    }
  }
  __syncthreads();

  for (int t = 0; t < T_; t++) {
    if (tid < 32) {
      // ---- prefetch raw x for t+1 (hidden behind h-WMMA / transcendentals) ----
      v4f nraw[16];
      if (t + 1 < T_) {
        const float* xr = inp + ((size_t)(t + 1) * B_ + m0 + an) * D_;
#pragma unroll
        for (int c = 0; c < 4; c++) {
          const float* p0 = xr + c * 32 + koA;
          const float* p1 = xr + c * 32 + 16 + koA;
          nraw[c * 4 + 0] = *(const v4f*)p0;
          nraw[c * 4 + 1] = *(const v4f*)(p0 + 4);
          nraw[c * 4 + 2] = *(const v4f*)p1;
          nraw[c * 4 + 3] = *(const v4f*)(p1 + 4);
        }
        if (t + 2 < T_) {
          const char* pf = (const char*)(inp + ((size_t)(t + 2) * B_ + m0 + an) * D_)
                           + (lane >> 4) * 256;
          __builtin_prefetch(pf, 0, 3);  // global_prefetch_b8, near-cache temporal
        }
      }

      // ---- critical path: h-projection only (x part was pipelined) ------------
      v8f acc0 = xacc0, acc1 = xacc1;
      union { v16h v; v8h h[2]; } af[4];
#pragma unroll
      for (int c = 0; c < 4; c++) {                  // issue all 8 ds_load_b128 first
        af[c].h[0] = *(const v8h*)(lds_h + an * HSTR + c * 32 + koA);
        af[c].h[1] = *(const v8h*)(lds_h + an * HSTR + c * 32 + 16 + koA);
      }
#pragma unroll
      for (int c = 0; c < 4; c++) {
        acc0 = __builtin_amdgcn_wmma_f32_16x16x32_f16(false, af[c].v, false, wbh[0][c],
                                                      (short)0, acc0, false, false);
        acc1 = __builtin_amdgcn_wmma_f32_16x16x32_f16(false, af[c].v, false, wbh[1][c],
                                                      (short)0, acc1, false, false);
      }

      // ---- sin^2(z/2) into LDS scratch (C layout: lane=n, vgpr=m row) ---------
#pragma unroll
      for (int v = 0; v < 8; v++) {
        int row = (lane < 16) ? v : v + 8;           // sample index
        float s0 = __sinf(0.5f * acc0[v]);
        float s1 = __sinf(0.5f * acc1[v]);
        zbuf[row * ZSTR + an]      = s0 * s0;        // n = 0..15  (f|i)
        zbuf[row * ZSTR + 16 + an] = s1 * s1;        // n = 16..31 (u|o)
      }

      // ---- per-sample reduce over 8 qubits -> 4 gate scalars ------------------
      {
        int m  = an;
        int cb = (lane < 16) ? 0 : 16;               // lanes 0-15: f,i ; 16-31: u,o
        v4f r0 = *(const v4f*)(zbuf + m * ZSTR + cb);
        v4f r1 = *(const v4f*)(zbuf + m * ZSTR + cb + 4);
        v4f r2 = *(const v4f*)(zbuf + m * ZSTR + cb + 8);
        v4f r3 = *(const v4f*)(zbuf + m * ZSTR + cb + 12);
        float sA = ((r0[0] + r0[1]) + (r0[2] + r0[3])) + ((r1[0] + r1[1]) + (r1[2] + r1[3]));
        float sB = ((r2[0] + r2[1]) + (r2[2] + r2[3])) + ((r3[0] + r3[1]) + (r3[2] + r3[3]));
        sA *= 0.125f;  sB *= 0.125f;                 // mean over NQ=8
        if (lane < 16) {
          gbuf[m]      = sigm(sA);                   // f
          gbuf[16 + m] = sigm(sB);                   // i
        } else {
          gbuf[32 + m] = tanh_fast(sA);              // u (g)
          gbuf[48 + m] = sigm(sB);                   // o
        }
      }

      // ---- pipeline: x-projection for t+1 (overlaps elementwise phase) --------
      if (t + 1 < T_) {
        v16h xf[4];
#pragma unroll
        for (int c = 0; c < 4; c++)
          xf[c] = cvt_frag(nraw[c * 4 + 0], nraw[c * 4 + 1],
                           nraw[c * 4 + 2], nraw[c * 4 + 3]);
#pragma unroll
        for (int v = 0; v < 8; v++) { xacc0[v] = bias0; xacc1[v] = bias1; }
#pragma unroll
        for (int c = 0; c < 4; c++) {
          xacc0 = __builtin_amdgcn_wmma_f32_16x16x32_f16(false, xf[c], false, wbx[0][c],
                                                         (short)0, xacc0, false, false);
          xacc1 = __builtin_amdgcn_wmma_f32_16x16x32_f16(false, xf[c], false, wbx[1][c],
                                                         (short)0, xacc1, false, false);
        }
      }
    }
    __syncthreads();  // gates visible to all waves; wave0 h-reads done before h writes

    // ---- broadcast LSTM update, all 128 threads: c,h over [16 samples][128] ---
    {
      float fg = gbuf[em];
      float ig = gbuf[16 + em];
      float ug = gbuf[32 + em];
      float og = gbuf[48 + em];
      float hv[16];
#pragma unroll
      for (int i = 0; i < 16; i++) {
        float cn = fg * creg[i] + ig * ug;
        creg[i] = cn;
        hv[i] = og * tanh_fast(cn);
      }
      float* orow = outStack + ((size_t)t * B_ + m0 + em) * H_ + ej;
#pragma unroll
      for (int i = 0; i < 4; i++) {
        v4f hq = { hv[4 * i], hv[4 * i + 1], hv[4 * i + 2], hv[4 * i + 3] };
        *(v4f*)(orow + 4 * i) = hq;
      }
      // repack h as f16 into LDS for next step's WMMA A operand
      _Float16* hl = lds_h + em * HSTR + ej;
      v8h p0, p1;
#pragma unroll
      for (int i = 0; i < 8; i++) { p0[i] = (_Float16)hv[i]; p1[i] = (_Float16)hv[8 + i]; }
      *(v8h*)(hl)     = p0;
      *(v8h*)(hl + 8) = p1;

      if (t == T_ - 1) {  // final hx / cx tails of d_out
        float* hxr = outHx + (size_t)(m0 + em) * H_ + ej;
        float* cxr = outCx + (size_t)(m0 + em) * H_ + ej;
#pragma unroll
        for (int i = 0; i < 4; i++) {
          v4f hq = { hv[4 * i], hv[4 * i + 1], hv[4 * i + 2], hv[4 * i + 3] };
          v4f cq = { creg[4 * i], creg[4 * i + 1], creg[4 * i + 2], creg[4 * i + 3] };
          *(v4f*)(hxr + 4 * i) = hq;
          *(v4f*)(cxr + 4 * i) = cq;
        }
      }
    }
    __syncthreads();  // h buffer complete before next step's wave0 ds_loads
  }
}

extern "C" void kernel_launch(void* const* d_in, const int* in_sizes, int n_in,
                              void* d_out, int out_size, void* d_ws, size_t ws_size,
                              hipStream_t stream) {
  (void)in_sizes; (void)n_in; (void)out_size; (void)d_ws; (void)ws_size;
  const float* inp = (const float*)d_in[0];
  const float* Wf  = (const float*)d_in[1];
  const float* bf  = (const float*)d_in[2];
  const float* Wi  = (const float*)d_in[3];
  const float* bi  = (const float*)d_in[4];
  const float* Wu  = (const float*)d_in[5];
  const float* bu  = (const float*)d_in[6];
  const float* Wo  = (const float*)d_in[7];
  const float* bo  = (const float*)d_in[8];
  qlstm_scan_kernel<<<dim3(B_ / 16), dim3(128), 0, stream>>>(
      inp, Wf, bf, Wi, bi, Wu, bu, Wo, bo, (float*)d_out);
}
